// LFA_44573170598615
// MI455X (gfx1250) — matrix operand order repaired and checked by reading, not compiled, not run on hardware
//
#include <hip/hip_runtime.h>
#include <hip/hip_bf16.h>

typedef __attribute__((ext_vector_type(16))) _Float16 v16h;
typedef __attribute__((ext_vector_type(8)))  float    v8f;

struct u128 { unsigned int w[4]; };
struct F4   { float f[4]; };

union U16h { v16h v; _Float16 h[16]; u128 q[2]; };
union U8f  { v8f  v; float     f[8];  u128 q[2]; };
union U8h  { _Float16 h[8]; u128 q; };

#ifndef __has_builtin
#define __has_builtin(x) 0
#endif

#if __has_builtin(__builtin_amdgcn_tensor_load_to_lds)
#define HAVE_TDM 1
#else
#define HAVE_TDM 0
#endif

#if __has_builtin(__builtin_amdgcn_s_wait_tensorcnt)
#define WAIT_TENSORCNT0() __builtin_amdgcn_s_wait_tensorcnt(0)
#else
#define WAIT_TENSORCNT0() asm volatile("s_wait_tensorcnt 0x0" ::: "memory")
#endif

// ---------------- ws layout (bytes, all 256-aligned) ----------------
// Aws : packed f16 A-fragments for W_hq/W_mk/W_mv : 3*4*2*32*16 f16 = 24576 B
// Qws : [b][y][x][c]      f16  4*128*128*64  = 8,388,608 B   @ 32768
// Kp  : [b][yp][xp][c]    f16  4*134*134*64  = 9,193,472 B   @ 8421376
// Vp  : [b][yp][c][xp144] f16  4*134*64*144  = 9,879,552 B   @ 17614848
#define AWS_OFF 0
#define QWS_OFF 32768
#define KP_OFF  8421376
#define VP_OFF  17614848

// ------------------------------------------------------------------
// Pack 64x64 weights into WMMA A-fragment per-lane order (f16).
// A 16x32 f16 layout: lane L: M = L%16, K = (e<8?e:e+8) + (L>=16?8:0)
// ------------------------------------------------------------------
__global__ void lfa_pack_w(const float* __restrict__ Whq,
                           const float* __restrict__ Wmk,
                           const float* __restrict__ Wmv,
                           _Float16* __restrict__ Aws) {
  int tid = blockIdx.x * blockDim.x + threadIdx.x;
  if (tid >= 3 * 4 * 2 * 32) return;
  int lane = tid & 31;
  int s    = (tid >> 5) & 1;   // K-step (32 in-ch)
  int mt   = (tid >> 6) & 3;   // 16 out-ch tile
  int w    = tid >> 8;         // which matrix
  const float* W = (w == 0) ? Whq : ((w == 1) ? Wmk : Wmv);
  int o = mt * 16 + (lane & 15);
  int koff = (lane >= 16) ? 8 : 0;
  long off = (long)tid * 16;
#pragma unroll
  for (int e = 0; e < 16; ++e) {
    int c = s * 32 + ((e < 8) ? e : (e + 8)) + koff;
    Aws[off + e] = (_Float16)W[o * 64 + c];
  }
}

// ------------------------------------------------------------------
// Zero only the pad ring / alignment slack of Kp and Vp.
//  A: Kp pad pixels (per b: 6 full rows of 134 + 128 rows x 6 cols = 1572 px),
//     each = 64 ch f16 = 8 u128 chunks                  -> 4*1572*8  = 50304
//  B: Vp full pad rows (per b: 6 rows of 64*144 f16)    -> 4*6*1152  = 27648 u128
//  C: Vp interior ring: xp in {0..2, 131..143} per ch   -> 4*128*64*16 = 524288 f16
// ------------------------------------------------------------------
__global__ void lfa_zero_pad(_Float16* __restrict__ Kp,
                             _Float16* __restrict__ Vp) {
  int tid = blockIdx.x * blockDim.x + threadIdx.x;
  u128 z; z.w[0] = 0u; z.w[1] = 0u; z.w[2] = 0u; z.w[3] = 0u;
  if (tid < 50304) {
    int c = tid & 7;
    int p = tid >> 3;                  // [0, 6288)
    int b = p / 1572;
    int r = p - b * 1572;
    int yp, xp;
    if (r < 402)      { yp = r / 134;                xp = r % 134; }
    else if (r < 804) { int q = r - 402; yp = 131 + q / 134; xp = q % 134; }
    else              { int q = r - 804; yp = 3 + q / 6;
                        int s6 = q % 6; xp = (s6 < 3) ? s6 : 128 + s6; }
    *(u128*)(Kp + (((b * 134 + yp) * 134) + xp) * 64 + c * 8) = z;
    return;
  }
  tid -= 50304;
  if (tid < 27648) {
    int c = tid % 1152;
    int p = tid / 1152;                // [0, 24)
    int b = p / 6, rr = p % 6;
    int yp = (rr < 3) ? rr : 128 + rr; // 0,1,2,131,132,133
    *(u128*)(Vp + (long)(b * 134 + yp) * 64 * 144 + c * 8) = z;
    return;
  }
  tid -= 27648;
  if (tid < 524288) {
    int s16 = tid & 15;
    int q   = tid >> 4;                // [0, 32768)
    int ch  = q & 63;
    int q2  = q >> 6;                  // [0, 512)
    int yv  = q2 & 127, b = q2 >> 7;
    int xp  = (s16 < 3) ? s16 : 128 + s16;   // 0..2, 131..143
    Vp[((b * 134 + yv + 3) * 64 + ch) * 144 + xp] = (_Float16)0.0f;
  }
}

#if HAVE_TDM
typedef __attribute__((ext_vector_type(4))) unsigned int u32x4;
typedef __attribute__((ext_vector_type(8))) int i32x8;
typedef __attribute__((ext_vector_type(4))) int i32x4;

// TDM: load a 64x64 f32 tile (line stride 16384 elems) into LDS with a
// 4-DWORD pad after every 64 DWORDs -> LDS rows of 68 floats.
__device__ __forceinline__ void tdm_load_tile(const float* gptr, unsigned lds_addr) {
  unsigned long long ga = (unsigned long long)gptr;
  // group0: count=1 | lds_addr | global_addr[56:0] | type=2 (bits 127:126)
  u32x4 g0 = { 1u, lds_addr, (unsigned)(ga & 0xffffffffu),
               ((unsigned)((ga >> 32) & 0x01ffffffu)) | 0x80000000u };
  // group1: data_size=4B, pad_enable, pad_interval=64DW(5), pad_amount=4DW(3);
  // tensor_dim0=64, tensor_dim1=64, tile_dim0=64, tile_dim1=64, dim0_stride=16384
  i32x8 g1 = { (int)((2u << 16) | (1u << 20) | (5u << 22) | (3u << 25)),
               (64 << 16),   // d1: tensor_dim0[15:0] at bits 63:48
               (64 << 16),   // d2: tensor_dim1[15:0] at bits 95:80
               (64 << 16),   // d3: tile_dim0 at bits 127:112
               64,           // d4: tile_dim1 at bits 143:128
               16384,        // d5: tensor_dim0_stride[31:0]
               0, 0 };
  i32x4 gz = { 0, 0, 0, 0 };
#if __clang_major__ >= 23
  i32x8 gz8 = { 0, 0, 0, 0, 0, 0, 0, 0 };
  __builtin_amdgcn_tensor_load_to_lds(g0, g1, gz, gz, gz8, 0);
#else
  __builtin_amdgcn_tensor_load_to_lds(g0, g1, gz, gz, 0);
#endif
}
#endif

// ------------------------------------------------------------------
// QKV projection. Block = 4 waves = 64 consecutive pixels of one row.
// Stage H/M tile (64ch x 64px f32) into LDS [ch][68], then per-wave
// WMMA: D[16 out-ch x 16 px] = Wtile[16x32] * Xtile[32x16], 2 K-steps.
// B 32x16 f16 layout: lane L: K = (e<8?e:e+8) + (L>=16?8:0), N = L%16
// ------------------------------------------------------------------
__global__ void __launch_bounds__(128) lfa_qkv(
    const float* __restrict__ H, const float* __restrict__ M,
    const _Float16* __restrict__ Aws,
    _Float16* __restrict__ Qws, _Float16* __restrict__ Kp,
    _Float16* __restrict__ Vp) {
  __shared__ float sH[64][68];   // row = in-ch, col = block-local px (+4 pad)
  __shared__ float sM[64][68];

  int t = threadIdx.x;
  int lane = t & 31;
  int wave = t >> 5;
  int blk = blockIdx.x;                    // 0..1023
  int b = blk >> 8;
  int y = (blk >> 1) & 127;
  int xblk = (blk & 1) * 64;
  int gbase = ((b * 64) * 128 + y) * 128 + xblk;   // ch = 0

#if HAVE_TDM
  if (wave == 0) {
    tdm_load_tile(H + gbase, (unsigned)(unsigned long long)&sH[0][0]);
    tdm_load_tile(M + gbase, (unsigned)(unsigned long long)&sM[0][0]);
    WAIT_TENSORCNT0();
  }
#else
  // manual coalesced staging: 1024 chunks of 16B per input
#pragma unroll
  for (int k = 0; k < 8; ++k) {
    int chunk = t + k * 128;               // 0..1023
    int ch = chunk >> 4;
    int xq = (chunk & 15) * 4;
    int gi = gbase + ch * 16384 + xq;
    F4 vh = *(const F4*)(H + gi);
    F4 vm = *(const F4*)(M + gi);
    *(F4*)&sH[ch][xq] = vh;                // rows are 272B = 16B-aligned
    *(F4*)&sM[ch][xq] = vm;
  }
#endif
  __syncthreads();

  int lo   = lane & 15;
  int hi8  = (lane >= 16) ? 8 : 0;
  int hi16 = (lane >= 16) ? 16 : 0;
  int pxl  = wave * 16 + lo;               // block-local pixel
  int x0   = xblk + wave * 16;

  // Build B fragments from LDS (f32 -> f16)
  U16h bh[2], bm[2];
#pragma unroll
  for (int s = 0; s < 2; ++s) {
#pragma unroll
    for (int e = 0; e < 16; ++e) {
      int ch = s * 32 + ((e < 8) ? e : (e + 8)) + hi8;
      bh[s].h[e] = (_Float16)sH[ch][pxl];
      bm[s].h[e] = (_Float16)sM[ch][pxl];
    }
  }

  const v16h* Av = (const v16h*)Aws;

  // ---- Q = W_hq @ H -> Qws [b][y][x][c] ----
#pragma unroll
  for (int mt = 0; mt < 4; ++mt) {
    v8f acc = {};
#pragma unroll
    for (int s = 0; s < 2; ++s) {
      v16h a = Av[((0 * 4 + mt) * 2 + s) * 32 + lane];
      acc = __builtin_amdgcn_wmma_f32_16x16x32_f16(false, a, false, bh[s].v,
                                                   (short)0, acc, false, false);
    }
    U8f r; r.v = acc;
    U8h pk;
#pragma unroll
    for (int g = 0; g < 8; ++g) pk.h[g] = (_Float16)r.f[g];
    int ch0 = mt * 16 + hi8;               // D: M = g + hi8 (rel out-ch)
    int qidx = (((b * 128 + y) * 128) + x0 + lo) * 64 + ch0;
    *(u128*)(Qws + qidx) = pk.q;
  }

  int yp = y + 3, xp = x0 + lo + 3;

  // ---- K = W_mk @ M -> Kp [b][yp][xp][c] ----
#pragma unroll
  for (int mt = 0; mt < 4; ++mt) {
    v8f acc = {};
#pragma unroll
    for (int s = 0; s < 2; ++s) {
      v16h a = Av[((1 * 4 + mt) * 2 + s) * 32 + lane];
      acc = __builtin_amdgcn_wmma_f32_16x16x32_f16(false, a, false, bm[s].v,
                                                   (short)0, acc, false, false);
    }
    U8f r; r.v = acc;
    U8h pk;
#pragma unroll
    for (int g = 0; g < 8; ++g) pk.h[g] = (_Float16)r.f[g];
    int ch0 = mt * 16 + hi8;
    int kidx = (((b * 134 + yp) * 134) + xp) * 64 + ch0;
    *(u128*)(Kp + kidx) = pk.q;
  }

  // ---- V = W_mv @ M -> Vp [b][yp][c][xp] (row stride 144) ----
#pragma unroll
  for (int mt = 0; mt < 4; ++mt) {
    v8f acc = {};
#pragma unroll
    for (int s = 0; s < 2; ++s) {
      v16h a = Av[((2 * 4 + mt) * 2 + s) * 32 + lane];
      acc = __builtin_amdgcn_wmma_f32_16x16x32_f16(false, a, false, bm[s].v,
                                                   (short)0, acc, false, false);
    }
    U8f r; r.v = acc;
#pragma unroll
    for (int g = 0; g < 8; ++g) {
      int ch = mt * 16 + hi8 + g;
      Vp[((b * 134 + yp) * 64 + ch) * 144 + xp] = (_Float16)r.f[g];
    }
  }
}

// ------------------------------------------------------------------
// Fused attention: banded WMMA scores -> softmax(49) -> banded WMMA out.
// One wave per 16-pixel row segment; 4 waves/block.
// ------------------------------------------------------------------
__global__ void __launch_bounds__(128) lfa_attn(
    const _Float16* __restrict__ Qws, const _Float16* __restrict__ Kp,
    const _Float16* __restrict__ Vp, float* __restrict__ out) {
  __shared__ float s_sc[4][16][56];       // scores / exp values, k = i*7+j
  __shared__ float s_red[4][2][2][16];    // [wave][max|sum][half][px]
  __shared__ float s_inv[4][16];

  int lane = threadIdx.x & 31;
  int wave = threadIdx.x >> 5;
  int tile = blockIdx.x * 4 + wave;
  int xt = tile & 7;
  int y  = (tile >> 3) & 127;
  int b  = tile >> 10;
  int x0 = xt * 16;
  int lo   = lane & 15;
  bool hi  = lane >= 16;
  int hi8  = hi ? 8 : 0;
  int hi16 = hi ? 16 : 0;

  // A fragments of Q: A[px][ch], px = L%16, K = e + hi16 (+s*32)
  U16h aq[2];
  int qbase = (((b * 128 + y) * 128) + x0 + lo) * 64;
#pragma unroll
  for (int s = 0; s < 2; ++s) {
    int c0 = s * 32 + hi16;
    aq[s].q[0] = *(const u128*)(Qws + qbase + c0);
    aq[s].q[1] = *(const u128*)(Qws + qbase + c0 + 8);
  }

  // -------- Phase 1: banded scores --------
#pragma unroll
  for (int i = 0; i < 7; ++i) {
    int yp = y + i;
    if (i < 6)  // prefetch next K row (gfx1250 global_prefetch)
      __builtin_prefetch(Kp + ((b * 134 + yp + 1) * 134 + x0 + lo) * 64 + hi16, 0, 1);
#pragma unroll
    for (int t = 0; t < 2; ++t) {
      int u0 = t * 6;                      // N-tile covers u in [u0, u0+15]
      v8f sa = {};
#pragma unroll
      for (int s = 0; s < 2; ++s) {
        // B[ch][u]: lane L: N = u0+lo (pixel column), K = e + hi16 + s*32
        const v16h* bp = (const v16h*)(
            Kp + ((b * 134 + yp) * 134 + x0 + u0 + lo) * 64 + hi16 + s * 32);
        sa = __builtin_amdgcn_wmma_f32_16x16x32_f16(false, aq[s].v, false, *bp,
                                                    (short)0, sa, false, false);
      }
      U8f r; r.v = sa;
#pragma unroll
      for (int g = 0; g < 8; ++g) {
        int px = g + hi8;
        int u  = u0 + lo;
        int j  = u - px;
        bool ok = (j >= 0) && (j <= 6) && ((t == 0) ? (u < 16) : (u >= 16));
        if (ok) {
          int di = i - 3, dj = j - 3;
          float w = (di == 0 && dj == 0)
                        ? 0.0f
                        : __expf(-(float)(di * di + dj * dj) * (1.0f / 36.0f));
          s_sc[wave][px][i * 7 + j] = r.f[g] * w;
        }
      }
    }
  }
  __syncthreads();

  // -------- Phase 2: softmax over 49 per pixel --------
  {
    int px = lo, half = hi ? 1 : 0;
    int k0 = half ? 25 : 0, k1 = half ? 49 : 25;
    float m = -3.0e38f;
    for (int k = k0; k < k1; ++k) m = fmaxf(m, s_sc[wave][px][k]);
    s_red[wave][0][half][px] = m;
  }
  __syncthreads();
  {
    int px = lo, half = hi ? 1 : 0;
    float m = fmaxf(s_red[wave][0][0][px], s_red[wave][0][1][px]);
    int k0 = half ? 25 : 0, k1 = half ? 49 : 25;
    float sum = 0.0f;
    for (int k = k0; k < k1; ++k) {
      float e = __expf(s_sc[wave][px][k] - m);
      s_sc[wave][px][k] = e;
      sum += e;
    }
    s_red[wave][1][half][px] = sum;
  }
  __syncthreads();
  if (!hi) s_inv[wave][lo] = 1.0f / (s_red[wave][1][0][lo] + s_red[wave][1][1][lo]);
  __syncthreads();

  // -------- Phase 3: out[px][ch] = sum_u alpha_band[px][u] * V[ch][x0+u] --------
  v8f zero = {};
  v8f oacc[4];
#pragma unroll
  for (int ct = 0; ct < 4; ++ct) oacc[ct] = zero;

  float inv = s_inv[wave][lo];
#pragma unroll
  for (int i = 0; i < 7; ++i) {
    int yp = y + i;
    if (i < 6)  // prefetch next V row
      __builtin_prefetch(Vp + ((b * 134 + yp + 1) * 64 + lo) * 144 + x0 + hi16, 0, 1);
    // banded alpha A fragment: A[px=lo][u], K = e + hi16
    U16h aa;
#pragma unroll
    for (int e = 0; e < 16; ++e) {
      int u = e + hi16;
      int j = u - lo;
      aa.h[e] = (j >= 0 && j <= 6)
                    ? (_Float16)(s_sc[wave][lo][i * 7 + j] * inv)
                    : (_Float16)0.0f;
    }
#pragma unroll
    for (int ct = 0; ct < 4; ++ct) {
      // B[u][ch]: lane L: N = ct*16+lo (channel), K = u = e + hi16
      const v16h* bv = (const v16h*)(
          Vp + ((b * 134 + yp) * 64 + ct * 16 + lo) * 144 + x0 + hi16);
      oacc[ct] = __builtin_amdgcn_wmma_f32_16x16x32_f16(false, aa.v, false, *bv,
                                                        (short)0, oacc[ct],
                                                        false, false);
    }
  }

  // Store: D[px][ch]; lane holds ch = ct*16+lo, px = g+hi8 -> 8 consecutive x
#pragma unroll
  for (int ct = 0; ct < 4; ++ct) {
    U8f r; r.v = oacc[ct];
    int ch = ct * 16 + lo;
    int obase = (((b * 64 + ch) * 128) + y) * 128 + x0 + hi8;
    *(u128*)(out + obase)     = r.q[0];
    *(u128*)(out + obase + 4) = r.q[1];
  }
}

// ------------------------------------------------------------------
extern "C" void kernel_launch(void* const* d_in, const int* in_sizes, int n_in,
                              void* d_out, int out_size, void* d_ws,
                              size_t ws_size, hipStream_t stream) {
  (void)in_sizes; (void)n_in; (void)out_size; (void)ws_size;
  const float* H   = (const float*)d_in[0];
  const float* M   = (const float*)d_in[1];
  const float* Whq = (const float*)d_in[2];
  const float* Wmk = (const float*)d_in[3];
  const float* Wmv = (const float*)d_in[4];
  float* out = (float*)d_out;

  char* wsb = (char*)d_ws;
  _Float16* Aws = (_Float16*)(wsb + AWS_OFF);
  _Float16* Qws = (_Float16*)(wsb + QWS_OFF);
  _Float16* Kp  = (_Float16*)(wsb + KP_OFF);
  _Float16* Vp  = (_Float16*)(wsb + VP_OFF);

  // 1) pack weights into WMMA A-fragment order
  lfa_pack_w<<<3, 256, 0, stream>>>(Whq, Wmk, Wmv, Aws);
  // 2) zero only the pad ring of the padded K/V buffers
  lfa_zero_pad<<<2353, 256, 0, stream>>>(Kp, Vp);
  // 3) QKV projections (TDM/LDS staged, WMMA)
  lfa_qkv<<<1024, 128, 0, stream>>>(H, M, Aws, Qws, Kp, Vp);
  // 4) fused local attention (WMMA scores + softmax + WMMA aggregation)
  lfa_attn<<<1024, 128, 0, stream>>>(Qws, Kp, Vp, out);
}
